// multi_DG_61735859913586
// MI455X (gfx1250) — matrix operand order repaired and checked by reading, not compiled, not run on hardware
//
#include <hip/hip_runtime.h>

typedef __attribute__((ext_vector_type(16))) _Float16 v16h;
typedef __attribute__((ext_vector_type(8)))  _Float16 v8h;
typedef __attribute__((ext_vector_type(8)))  float    v8f;

constexpr int NROWS   = 65536;
constexpr int DIN     = 495;
constexpr int KP      = 512;   // padded K for input projection
constexpr int MODELS  = 24;
constexpr int HID     = 64;
constexpr int NLAYERS = 5;
constexpr int RT      = 2;     // 16-row tiles per wave (register M-blocking: B reused RT times)

union Frag16 { v16h v; v8h h[2]; };

// ---- prep: x (f32 [N][495]) -> f16 [N][512], zero padded ----
__global__ void cvt_x(const float* __restrict__ x, _Float16* __restrict__ xh) {
  int idx = blockIdx.x * blockDim.x + threadIdx.x;   // over NROWS*KP (exact)
  int row = idx >> 9, k = idx & 511;
  float v = (k < DIN) ? x[row * DIN + k] : 0.f;
  xh[idx] = (_Float16)v;
}

// ---- prep: W_in (f32 [m][495][64]) -> f16 transposed [m][64][512], zero padded ----
__global__ void cvt_win(const float* __restrict__ w, _Float16* __restrict__ wt) {
  int idx = blockIdx.x * blockDim.x + threadIdx.x;   // over MODELS*HID*KP (exact)
  int k = idx & 511; int t = idx >> 9; int h = t & 63; int m = t >> 6;
  float v = (k < DIN) ? w[(m * DIN + k) * HID + h] : 0.f;
  wt[idx] = (_Float16)v;
}

// ---- prep: W_h (f32 [m][l][64in][64out]) -> f16 transposed [m][l][64out][64in] ----
__global__ void cvt_wh(const float* __restrict__ w, _Float16* __restrict__ wt) {
  int idx = blockIdx.x * blockDim.x + threadIdx.x;   // over MODELS*NLAYERS*HID*HID (exact)
  int hin = idx & 63; int t = idx >> 6; int hout = t & 63; int ml = t >> 6;
  wt[idx] = (_Float16)w[(ml * HID + hin) * HID + hout];
}

// ---- main fused kernel: one model x 256-row block per workgroup, 8 waves x 32 rows ----
__launch_bounds__(256)
__global__ void dg_main(const _Float16* __restrict__ xh,
                        const _Float16* __restrict__ wint,  // [m][64][512]
                        const _Float16* __restrict__ wht,   // [m][l][64][64]
                        const float* __restrict__ b_in,     // [m][64]
                        const float* __restrict__ b_h,      // [m][l][64]
                        const float* __restrict__ w_out,    // [m][64]
                        const float* __restrict__ b_out,    // [m]
                        float* __restrict__ out) {          // [N][24]
  // wave-private activation tiles; stride 72 keeps b128 reads 16B aligned + bank-spread
  __shared__ __align__(16) _Float16 hbuf[8][RT][16][72];
  // wave-private f16 copy of h0 for the residual (frees 64 f32 VGPRs -> higher occupancy)
  __shared__ __align__(16) _Float16 h0buf[8][RT][16][66];

  const int m    = blockIdx.y;
  const int wave = threadIdx.x >> 5;
  const int lane = threadIdx.x & 31;
  const int r16  = lane & 15;   // N index within 16-col tile / A row
  const int hs   = lane >> 4;   // half-select (K phase for A/B, +8 rows for C/D)
  const int baseRow = blockIdx.x * (128 * RT) + wave * (16 * RT);

  // ---------- input projection: h0 = x_tile[32x512] @ W_in[m] (512x64) ----------
  v8f acc[RT][4];
  #pragma unroll
  for (int rt = 0; rt < RT; rt++)
    #pragma unroll
    for (int nt = 0; nt < 4; nt++)
      #pragma unroll
      for (int v = 0; v < 8; v++) acc[rt][nt][v] = 0.f;

  const _Float16* arow0 = xh + (size_t)(baseRow + r16) * KP + hs * 8;
  const _Float16* wbase = wint + (size_t)m * HID * KP;

  for (int ks = 0; ks < 16; ks++) {            // K = 512, 32 per step
    Frag16 a[RT];
    #pragma unroll
    for (int rt = 0; rt < RT; rt++) {
      const _Float16* ap = arow0 + (size_t)(rt * 16) * KP + ks * 32;
      a[rt].h[0] = *(const v8h*)ap;            // K k0..k0+7   (or +8..15 on hi half)
      a[rt].h[1] = *(const v8h*)(ap + 16);     // K k0+16..23  (or +24..31)
    }
    #pragma unroll
    for (int nt = 0; nt < 4; nt++) {
      const _Float16* bp = wbase + (size_t)(nt * 16 + r16) * KP + ks * 32 + hs * 8;
      Frag16 b;
      b.h[0] = *(const v8h*)bp;
      b.h[1] = *(const v8h*)(bp + 16);
      #pragma unroll
      for (int rt = 0; rt < RT; rt++)          // reuse B across both row tiles
        acc[rt][nt] = __builtin_amdgcn_wmma_f32_16x16x32_f16(
            false, a[rt].v, false, b.v, (short)0, acc[rt][nt], false, false);
    }
  }

  // bias; spill f16 h0 to LDS twice: layer-0 A operand + residual copy
  #pragma unroll
  for (int rt = 0; rt < RT; rt++)
    #pragma unroll
    for (int nt = 0; nt < 4; nt++) {
      float bias = b_in[m * HID + nt * 16 + r16];
      #pragma unroll
      for (int v = 0; v < 8; v++) {
        float t = acc[rt][nt][v] + bias;
        _Float16 th = (_Float16)t;
        hbuf[wave][rt][v + 8 * hs][nt * 16 + r16]  = th;
        h0buf[wave][rt][v + 8 * hs][nt * 16 + r16] = th;
      }
    }

  // ---------- 5 hidden layers: h = relu(h @ W_h[l] + b_h[l]) ----------
  const _Float16* whm = wht + (size_t)m * NLAYERS * HID * HID;
  #pragma unroll 1
  for (int l = 0; l < NLAYERS; l++) {
    v8f a2[RT][4];
    #pragma unroll
    for (int rt = 0; rt < RT; rt++)
      #pragma unroll
      for (int nt = 0; nt < 4; nt++)
        #pragma unroll
        for (int v = 0; v < 8; v++) a2[rt][nt][v] = 0.f;

    #pragma unroll
    for (int ks = 0; ks < 2; ks++) {           // K = 64
      Frag16 a[RT];
      #pragma unroll
      for (int rt = 0; rt < RT; rt++) {
        const _Float16* lp = &hbuf[wave][rt][r16][ks * 32 + hs * 8];
        a[rt].h[0] = *(const v8h*)lp;
        a[rt].h[1] = *(const v8h*)(lp + 16);
      }
      #pragma unroll
      for (int nt = 0; nt < 4; nt++) {
        const _Float16* bp = whm + (size_t)(l * HID + nt * 16 + r16) * HID + ks * 32 + hs * 8;
        Frag16 b;
        b.h[0] = *(const v8h*)bp;
        b.h[1] = *(const v8h*)(bp + 16);
        #pragma unroll
        for (int rt = 0; rt < RT; rt++)
          a2[rt][nt] = __builtin_amdgcn_wmma_f32_16x16x32_f16(
              false, a[rt].v, false, b.v, (short)0, a2[rt][nt], false, false);
      }
    }

    #pragma unroll
    for (int rt = 0; rt < RT; rt++)
      #pragma unroll
      for (int nt = 0; nt < 4; nt++) {
        float bias = b_h[(m * NLAYERS + l) * HID + nt * 16 + r16];
        #pragma unroll
        for (int v = 0; v < 8; v++) {
          float t = a2[rt][nt][v] + bias;
          a2[rt][nt][v] = t > 0.f ? t : 0.f;   // relu
        }
      }

    if (l < NLAYERS - 1) {
      // stage next layer's A operand (same-wave LDS is in-order; no barrier needed)
      #pragma unroll
      for (int rt = 0; rt < RT; rt++)
        #pragma unroll
        for (int nt = 0; nt < 4; nt++)
          #pragma unroll
          for (int v = 0; v < 8; v++)
            hbuf[wave][rt][v + 8 * hs][nt * 16 + r16] = (_Float16)a2[rt][nt][v];
    } else {
      // residual add: h0 read back from LDS (f16)
      #pragma unroll
      for (int rt = 0; rt < RT; rt++)
        #pragma unroll
        for (int nt = 0; nt < 4; nt++)
          #pragma unroll
          for (int v = 0; v < 8; v++)
            acc[rt][nt][v] = a2[rt][nt][v] +
                (float)h0buf[wave][rt][v + 8 * hs][nt * 16 + r16];
    }
  }

  // ---------- head: out[row, m] = tanh(h . W_out[m] + b_out[m]) ----------
  float wo[4];
  #pragma unroll
  for (int nt = 0; nt < 4; nt++) wo[nt] = w_out[m * HID + nt * 16 + r16];
  const float bo = b_out[m];

  #pragma unroll
  for (int rt = 0; rt < RT; rt++) {
    float p[8];
    #pragma unroll
    for (int v = 0; v < 8; v++) p[v] = 0.f;
    #pragma unroll
    for (int nt = 0; nt < 4; nt++)
      #pragma unroll
      for (int v = 0; v < 8; v++) p[v] += acc[rt][nt][v] * wo[nt];
    // reduce over N: xor masks 1..8 only mix lanes within each 16-lane half (N = lane%16)
    #pragma unroll
    for (int off = 1; off < 16; off <<= 1)
      #pragma unroll
      for (int v = 0; v < 8; v++) p[v] += __shfl_xor(p[v], off, 32);

    if (r16 == 0) {  // lanes 0 (rows M=0..7) and 16 (rows M=8..15)
      #pragma unroll
      for (int v = 0; v < 8; v++) {
        int row = baseRow + rt * 16 + v + 8 * hs;
        out[(size_t)row * MODELS + m] = tanhf(p[v] + bo);
      }
    }
  }
}

extern "C" void kernel_launch(void* const* d_in, const int* in_sizes, int n_in,
                              void* d_out, int out_size, void* d_ws, size_t ws_size,
                              hipStream_t stream) {
  const float* x     = (const float*)d_in[0];
  const float* W_in  = (const float*)d_in[1];
  const float* b_in  = (const float*)d_in[2];
  const float* W_h   = (const float*)d_in[3];
  const float* b_h   = (const float*)d_in[4];
  const float* W_out = (const float*)d_in[5];
  const float* b_out = (const float*)d_in[6];
  float* out = (float*)d_out;

  char* ws = (char*)d_ws;
  _Float16* xh   = (_Float16*)ws;                                  // 65536*512*2   = 64 MiB
  _Float16* wint = (_Float16*)(ws + (size_t)NROWS * KP * 2);       // 24*64*512*2   = 1.5 MiB
  _Float16* wht  = (_Float16*)(ws + (size_t)NROWS * KP * 2
                                  + (size_t)MODELS * HID * KP * 2);// 24*5*64*64*2  = 0.94 MiB

  cvt_x  <<<(NROWS * KP) / 256,                   256, 0, stream>>>(x, xh);
  cvt_win<<<(MODELS * HID * KP) / 256,            256, 0, stream>>>(W_in, wint);
  cvt_wh <<<(MODELS * NLAYERS * HID * HID) / 256, 256, 0, stream>>>(W_h, wht);

  dim3 grid(NROWS / (128 * RT), MODELS);
  dg_main<<<grid, 256, 0, stream>>>(xh, wint, wht, b_in, b_h, W_out, b_out, out);
}
